// ValueNet_65850438582627
// MI455X (gfx1250) — compile-verified, matrix operand order checked by reading, not run on hardware
//
#include <hip/hip_runtime.h>
#include <hip/hip_bf16.h>

// ---------------- model dims ----------------
#define B_ 16
#define L_ 1024
#define OBS 64
#define H_ 256
#define NB 4
#define NST 16
#define DC 4
#define DIN 512            // EXP*H
#define DTR 16
#define BL (B_ * L_)       // 16384

// ---------------- GEMM tile ----------------
#define BM 64              // block rows   (4 row-tiles)
#define BN 64              // block cols   (2 col-pairs of 16x16 tiles per wave)
#define KT 32
#define LDP 40             // padded LDS row pitch (shorts): 80B, 16B-aligned, bank-friendly

typedef __attribute__((ext_vector_type(16))) __bf16          v16bf;
typedef __attribute__((ext_vector_type(8)))  float           v8f;
typedef __attribute__((ext_vector_type(8)))  unsigned short  us8;   // 16-byte LDS chunk

union FragU { v16bf v; us8 h[2]; unsigned short u[16]; };

__device__ __forceinline__ unsigned short f2bf(float f) {
    unsigned int x = __float_as_uint(f);
    unsigned int r = x + 0x7FFFu + ((x >> 16) & 1u);   // round-to-nearest-even
    return (unsigned short)(r >> 16);
}

__device__ __forceinline__ float softplus_f(float x) {
    return (x > 20.f) ? x : log1pf(expf(x));
}
__device__ __forceinline__ float silu_f(float x) {
    return x / (1.f + expf(-x));
}

// C[M,N] = A[M,K] * W[N,K]^T  (+bias, +softplus per mode)
// mode: 0 = none, 1 = +bias, 2 = softplus(x + bias)
// Requires: M % 64 == 0, K % 4 == 0, lda % 4 == 0, 16B-aligned base pointers.
// 256 threads = 8 waves; wave (wr,wc): rows 16*wr..+15, cols 32*wc..+31 (two 16x16 D tiles).
__global__ __launch_bounds__(256) void gemm_bf16_wmma(
    const float* __restrict__ A, int lda,
    const float* __restrict__ W,
    const float* __restrict__ bias,
    float* __restrict__ C,
    int M, int N, int K, int mode)
{
    __shared__ unsigned short As[BM][LDP];   // bf16 bits, [m][k]
    __shared__ unsigned short Bs[BN][LDP];   // bf16 bits, TRANSPOSED: Bs[n][k] = W[n][k]

    const int tid  = threadIdx.x;
    const int wave = __builtin_amdgcn_readfirstlane(tid >> 5);
    const int lane = tid & 31;
    const int wr   = wave >> 1;             // 0..3 -> row tile
    const int wc   = wave & 1;              // 0..1 -> col pair
    const int blockM = blockIdx.x * BM;
    const int blockN = blockIdx.y * BN;

    v8f acc0 = {0.f, 0.f, 0.f, 0.f, 0.f, 0.f, 0.f, 0.f};
    v8f acc1 = acc0;
    const int ksteps = (K + KT - 1) / KT;
    const bool act0 = (blockN + 32 * wc)      < N;   // wave-uniform tile guards
    const bool act1 = (blockN + 32 * wc + 16) < N;

    // fragment coordinates (constant across k-steps)
    const int fm   = 16 * wr + (lane & 15);
    const int fn0  = 32 * wc + (lane & 15);
    const int fn1  = fn0 + 16;
    const int half = lane >> 4;

    for (int ks = 0; ks < ksteps; ++ks) {
        const int k0 = ks * KT;

        // ---- stage A tile: 64x32 fp32 -> bf16, two float4 per thread ----
        #pragma unroll
        for (int it = 0; it < 2; ++it) {
            const int i  = tid + it * 256;     // 0..511
            const int r  = i >> 3;             // 0..63
            const int c4 = (i & 7) << 2;       // 0,4,...,28
            const int gk = k0 + c4;
            float4 v = make_float4(0.f, 0.f, 0.f, 0.f);
            if (gk < K)
                v = *reinterpret_cast<const float4*>(A + (size_t)(blockM + r) * lda + gk);
            ushort4 p;
            p.x = f2bf(v.x); p.y = f2bf(v.y); p.z = f2bf(v.z); p.w = f2bf(v.w);
            *reinterpret_cast<ushort4*>(&As[r][c4]) = p;
        }
        // ---- stage B tile transposed: Bs[n][k] = W[n][k], two float4 per thread ----
        #pragma unroll
        for (int it = 0; it < 2; ++it) {
            const int i  = tid + it * 256;     // 0..511
            const int nn = i >> 3;             // 0..63
            const int c4 = (i & 7) << 2;       // 0,4,...,28
            const int gn = blockN + nn, gk = k0 + c4;
            float4 v = make_float4(0.f, 0.f, 0.f, 0.f);
            if (gn < N && gk < K)
                v = *reinterpret_cast<const float4*>(W + (size_t)gn * K + gk);
            ushort4 p;
            p.x = f2bf(v.x); p.y = f2bf(v.y); p.z = f2bf(v.z); p.w = f2bf(v.w);
            *reinterpret_cast<ushort4*>(&Bs[nn][c4]) = p;
        }
        // prefetch next K slab of A (gfx1250 global_prefetch_b8)
        if (ks + 1 < ksteps) {
            __builtin_prefetch(A + (size_t)(blockM + (tid & 31)) * lda + k0 + KT, 0, 1);
        }
        __syncthreads();

        // CDNA5 16-bit fragment layout: lane-half h holds K = 8h..8h+7 and 16+8h..16+8h+7
        // -> two contiguous 16B LDS chunks per lane for A and for each B tile
        FragU a;
        a.h[0] = *reinterpret_cast<const us8*>(&As[fm][half << 3]);
        a.h[1] = *reinterpret_cast<const us8*>(&As[fm][16 + (half << 3)]);
        if (act0) {
            FragU b;
            b.h[0] = *reinterpret_cast<const us8*>(&Bs[fn0][half << 3]);
            b.h[1] = *reinterpret_cast<const us8*>(&Bs[fn0][16 + (half << 3)]);
            acc0 = __builtin_amdgcn_wmma_f32_16x16x32_bf16(
                false, a.v, false, b.v, (short)0, acc0, false, false);
        }
        if (act1) {
            FragU b;
            b.h[0] = *reinterpret_cast<const us8*>(&Bs[fn1][half << 3]);
            b.h[1] = *reinterpret_cast<const us8*>(&Bs[fn1][16 + (half << 3)]);
            acc1 = __builtin_amdgcn_wmma_f32_16x16x32_bf16(
                false, a.v, false, b.v, (short)0, acc1, false, false);
        }
        __syncthreads();
    }

    // epilogue: D layout: VGPR r -> M = r + 8*(lane/16), N = lane%16
    const int mrow = blockM + 16 * wr + 8 * (lane >> 4);
    #pragma unroll
    for (int t = 0; t < 2; ++t) {
        const bool act = t ? act1 : act0;
        if (!act) continue;
        const v8f& acc = t ? acc1 : acc0;
        const int n = blockN + 32 * wc + 16 * t + (lane & 15);
        if (n < N) {
            const float bv = (mode != 0) ? bias[n] : 0.f;
            #pragma unroll
            for (int r = 0; r < 8; ++r) {
                float v = acc[r] + bv;
                if (mode == 2) v = softplus_f(v);
                C[(size_t)(mrow + r) * N + n] = v;
            }
        }
    }
}

// causal depthwise conv (DC=4) + bias + SiLU; xin = first DIN cols of xz
__global__ __launch_bounds__(256) void conv_silu_kernel(
    const float* __restrict__ xz,   // [B,L,2*DIN]
    const float* __restrict__ cw,   // [DIN,DC]
    const float* __restrict__ cb,   // [DIN]
    float* __restrict__ u)          // [B,L,DIN]
{
    const int idx = blockIdx.x * blockDim.x + threadIdx.x;
    if (idx >= BL * DIN) return;
    const int d  = idx & (DIN - 1);
    const int bl = idx >> 9;            // /DIN
    const int l  = bl & (L_ - 1);
    const int b  = bl >> 10;            // /L_
    float s = cb[d];
    #pragma unroll
    for (int j = 0; j < DC; ++j) {
        const int ll = l - (DC - 1) + j;
        if (ll >= 0)
            s += xz[((size_t)b * L_ + ll) * (2 * DIN) + d] * cw[d * DC + j];
    }
    u[idx] = silu_f(s);
}

// selective scan: one thread per (b,d); NST=16 states in registers.
// fuses y = (scan + D*u) * silu(z)
__global__ __launch_bounds__(256) void scan_kernel(
    const float* __restrict__ dt,    // [B,L,DIN] (softplus'ed)
    const float* __restrict__ u,     // [B,L,DIN]
    const float* __restrict__ xp,    // [B,L,48]: [0:16)=dtl, [16:32)=Bc, [32:48)=Cc
    const float* __restrict__ xz,    // [B,L,2*DIN] (z = cols DIN..2*DIN)
    const float* __restrict__ Alog,  // [DIN,NST]
    const float* __restrict__ Dsk,   // [DIN]
    float* __restrict__ y)           // [B,L,DIN]
{
    const int idx = blockIdx.x * blockDim.x + threadIdx.x;
    if (idx >= B_ * DIN) return;
    const int d = idx & (DIN - 1);
    const int b = idx >> 9;

    float Ar[NST], h[NST];
    #pragma unroll
    for (int n = 0; n < NST; ++n) {
        Ar[n] = -expf(Alog[d * NST + n]);
        h[n]  = 0.f;
    }
    const float Dd = Dsk[d];

    for (int l = 0; l < L_; ++l) {
        const size_t base = (size_t)b * L_ + l;
        const float dtv = dt[base * DIN + d];
        const float uv  = u[base * DIN + d];
        const float zv  = xz[base * (2 * DIN) + DIN + d];
        const float* pr = xp + base * (DTR + 2 * NST);
        const float du  = dtv * uv;
        float accv = 0.f;
        #pragma unroll
        for (int n = 0; n < NST; ++n) {
            const float Bc = pr[DTR + n];
            const float Cc = pr[DTR + NST + n];
            h[n] = expf(dtv * Ar[n]) * h[n] + du * Bc;
            accv += h[n] * Cc;
        }
        y[base * DIN + d] = (accv + Dd * uv) * silu_f(zv);
    }
}

// v[b] = dot(h[b, L-1, :], head_w) + head_b  -- one block per b
__global__ __launch_bounds__(256) void head_kernel(
    const float* __restrict__ h, const float* __restrict__ hw,
    const float* __restrict__ hb, float* __restrict__ out)
{
    __shared__ float red[256];
    const int b = blockIdx.x, t = threadIdx.x;
    red[t] = h[((size_t)b * L_ + (L_ - 1)) * H_ + t] * hw[t];
    __syncthreads();
    for (int off = 128; off > 0; off >>= 1) {
        if (t < off) red[t] += red[t + off];
        __syncthreads();
    }
    if (t == 0) out[b] = red[0] + hb[0];
}

// ---------------- workspace layout (floats) ----------------
#define OFF_H   ((size_t)0)                       // [B,L,H]      4,194,304
#define OFF_XZ  (OFF_H  + (size_t)BL * H_)        // [B,L,2*DIN] 16,777,216
#define OFF_U   (OFF_XZ + (size_t)BL * 2 * DIN)   // [B,L,DIN]    8,388,608
#define OFF_XP  (OFF_U  + (size_t)BL * DIN)       // [B,L,48]       786,432
#define OFF_DT  (OFF_XP + (size_t)BL * 48)        // [B,L,DIN]    8,388,608
#define OFF_Y   (OFF_DT + (size_t)BL * DIN)       // [B,L,DIN]    8,388,608

extern "C" void kernel_launch(void* const* d_in, const int* in_sizes, int n_in,
                              void* d_out, int out_size, void* d_ws, size_t ws_size,
                              hipStream_t stream) {
    const float* x        = (const float*)d_in[0];   // [B,L,OBS]
    const float* proj_w   = (const float*)d_in[1];   // [H,OBS]
    const float* proj_b   = (const float*)d_in[2];   // [H]
    const float* in_pw    = (const float*)d_in[3];   // [NB,2*DIN,H]
    const float* conv_w   = (const float*)d_in[4];   // [NB,DIN,DC]
    const float* conv_b   = (const float*)d_in[5];   // [NB,DIN]
    const float* x_pw     = (const float*)d_in[6];   // [NB,48,DIN]
    const float* dt_pw    = (const float*)d_in[7];   // [NB,DIN,DTR]
    const float* dt_pb    = (const float*)d_in[8];   // [NB,DIN]
    const float* A_log    = (const float*)d_in[9];   // [NB,DIN,NST]
    const float* D_skip   = (const float*)d_in[10];  // [NB,DIN]
    const float* out_pw   = (const float*)d_in[11];  // [NB,H,DIN]
    const float* head_w   = (const float*)d_in[12];  // [1,H]
    const float* head_b   = (const float*)d_in[13];  // [1]
    float* out = (float*)d_out;

    float* ws = (float*)d_ws;
    float* h  = ws + OFF_H;
    float* xz = ws + OFF_XZ;
    float* u  = ws + OFF_U;
    float* xp = ws + OFF_XP;
    float* dt = ws + OFF_DT;
    float* y  = ws + OFF_Y;

    const dim3 blk(256);

    // h = x @ proj_w^T + proj_b : M=BL, K=OBS, N=H
    gemm_bf16_wmma<<<dim3(BL / BM, H_ / BN), blk, 0, stream>>>(
        x, OBS, proj_w, proj_b, h, BL, H_, OBS, 1);

    for (int i = 0; i < NB; ++i) {
        const float* ipw = in_pw  + (size_t)i * 2 * DIN * H_;
        const float* cw  = conv_w + (size_t)i * DIN * DC;
        const float* cb  = conv_b + (size_t)i * DIN;
        const float* xpw = x_pw   + (size_t)i * (DTR + 2 * NST) * DIN;
        const float* dpw = dt_pw  + (size_t)i * DIN * DTR;
        const float* dpb = dt_pb  + (size_t)i * DIN;
        const float* Al  = A_log  + (size_t)i * DIN * NST;
        const float* Dk  = D_skip + (size_t)i * DIN;
        const float* opw = out_pw + (size_t)i * H_ * DIN;

        // xz = h @ in_proj_w^T : M=BL, K=H, N=2*DIN
        gemm_bf16_wmma<<<dim3(BL / BM, (2 * DIN) / BN), blk, 0, stream>>>(
            h, H_, ipw, nullptr, xz, BL, 2 * DIN, H_, 0);

        // u = silu(causal_dwconv(xin) + cb)
        conv_silu_kernel<<<dim3((BL * DIN) / 256), blk, 0, stream>>>(xz, cw, cb, u);

        // xp = u @ x_proj_w^T : M=BL, K=DIN, N=48  (one 64-wide col block, guarded)
        gemm_bf16_wmma<<<dim3(BL / BM, 1), blk, 0, stream>>>(
            u, DIN, xpw, nullptr, xp, BL, DTR + 2 * NST, DIN, 0);

        // dt = softplus(xp[:, :16] @ dt_proj_w^T + dt_proj_b) : K=16 (lda=48)
        gemm_bf16_wmma<<<dim3(BL / BM, DIN / BN), blk, 0, stream>>>(
            xp, DTR + 2 * NST, dpw, dpb, dt, BL, DIN, DTR, 2);

        // selective scan + D-skip + silu(z) gate
        scan_kernel<<<dim3((B_ * DIN) / 256), blk, 0, stream>>>(
            dt, u, xp, xz, Al, Dk, y);

        // h = y @ out_proj_w^T : M=BL, K=DIN, N=H (overwrites h for next block)
        gemm_bf16_wmma<<<dim3(BL / BM, H_ / BN), blk, 0, stream>>>(
            y, DIN, opw, nullptr, h, BL, H_, DIN, 0);
    }

    head_kernel<<<dim3(B_), blk, 0, stream>>>(h, head_w, head_b, out);
}
